// MyConv1d_ell_2_sequential_64037962383414
// MI455X (gfx1250) — matrix operand order repaired and checked by reading, not compiled, not run on hardware
//
#include <hip/hip_runtime.h>
#include <hip/hip_bf16.h>

// ---------------------------------------------------------------------------
// Problem constants (from the reference)
// ---------------------------------------------------------------------------
#define NB    16384    // batch
#define NC    128      // in channels (K over c)
#define NO    128      // out channels
#define NLIN  18       // input length
#define NSLOT 54       // 9 t-slots * 6 lengths
#define NTAP  28       // merged (position,filter,k,mult) taps over all lengths
#define NP    15       // distinct x positions actually used: 0..14
#define WS_DW (NTAP * 4 * 8 * 32 * 8)   // 229376 dwords per hi/lo weight plane

typedef __attribute__((ext_vector_type(16))) __bf16 v16bf;
typedef __attribute__((ext_vector_type(8)))  float  v8f;

// Merged tap tables (derived from KK0/PAIRS in the reference):
//   d=0(len4): p{0,4} via f2(k0,k1)
//   d=1(len5): p{0,1} f2k0, p{5,6} f2k1
//   d=2(len6): p1 f2k0, p7 f2k1, p{0,4,8} f3(k0,k1,k2)
//   d=3(len7): p0 2*f3k0, p1 f3k0, p{4,5,6} f3k1, p9 f3k2, p10 2*f3k2
//   d=4(len8): p0 f3k0, p1 2*f3k0, p{5,6,7} f3k1, p11 2*f3k2, p12 f3k2
//   d=5(len9): p1 f3k0, p7 f3k1, p13 f3k2
constexpr int   TD[6]          = {6, 5, 4, 3, 2, 1};
constexpr int   TAP_START[7]   = {0, 2, 6, 11, 18, 25, 28};
constexpr int   TAP_P[NTAP]    = {0,4,  0,1,5,6,  1,7,0,4,8,
                                  0,1,4,5,6,9,10,  0,1,5,6,7,11,12,  1,7,13};
constexpr int   TAP_F[NTAP]    = {2,2,  2,2,2,2,  2,2,3,3,3,
                                  3,3,3,3,3,3,3,  3,3,3,3,3,3,3,  3,3,3};
constexpr int   TAP_K[NTAP]    = {0,1,  0,0,1,1,  0,1,0,1,2,
                                  0,0,1,1,1,2,2,  0,0,1,1,1,2,2,  0,1,2};
constexpr float TAP_M[NTAP]    = {1,1,  1,1,1,1,  1,1,1,1,1,
                                  2,1,1,1,1,1,2,  1,2,1,1,1,2,1,  1,1,1};

// 1/sqrt(2*C), 1/sqrt(3*C)
#define S2F 0.0625f
#define S3F 0.05103103630798288f

__device__ __forceinline__ unsigned int bf16_rne_bits(float f) {
  unsigned int u = __float_as_uint(f);
  return (u + 0x7FFFu + ((u >> 16) & 1u)) >> 16;
}
__device__ __forceinline__ void split2(float f, unsigned int& hi, unsigned int& lo) {
  hi = bf16_rne_bits(f);
  const float fh = __uint_as_float(hi << 16);
  lo = bf16_rne_bits(f - fh);
}

// ---------------------------------------------------------------------------
// Prep: bake taps into B-fragments (32x16 bf16, WMMA B layout), hi/lo planes.
// B order used here (32x16, contiguous k): lane = N + 16*(K>=16); VGPR v holds K = 16*(lane>=16)+2v
// ws dword index = (((j*4+q)*8+n)*32 + lane)*8 + v  == flat thread id here.
// ---------------------------------------------------------------------------
__global__ void prep_weights_kernel(const float* __restrict__ f2,
                                    const float* __restrict__ f3,
                                    unsigned int* __restrict__ wsHi,
                                    unsigned int* __restrict__ wsLo) {
  const int s = blockIdx.x * 256 + threadIdx.x;
  if (s >= WS_DW) return;
  const int v    = s & 7;
  const int lane = (s >> 3) & 31;
  const int n    = (s >> 8) & 7;    // o-tile
  const int q    = (s >> 11) & 3;   // c-chunk
  const int j    = s >> 13;         // tap
  const int o    = n * 16 + (lane & 15);
  const int c    = q * 32 + 16 * (lane >> 4) + 2 * v;

  const int   fsel = TAP_F[j];
  const int   kk   = TAP_K[j];
  const float sc   = TAP_M[j] * ((fsel == 2) ? S2F : S3F);
  const float* f   = (fsel == 2) ? f2 : f3;
  const int    kf  = (fsel == 2) ? 2 : 3;

  const float w0 = f[(o * NC + c) * kf + kk] * sc;
  const float w1 = f[(o * NC + c + 1) * kf + kk] * sc;
  unsigned int h0, l0, h1, l1;
  split2(w0, h0, l0);
  split2(w1, h1, l1);
  wsHi[s] = (h1 << 16) | h0;
  wsLo[s] = (l1 << 16) | l0;
}

// ---------------------------------------------------------------------------
// Main: one block = 16 batch rows; 8 waves = 8 o-tiles of 16.
// Phase 1: x -> LDS as A-fragments (16x32 bf16 WMMA A layout), hi/lo planes.
// A order used here (16x32, contiguous k): lane = M + 16*half; lanes0-15: K 0..15 (v0..7);
// lanes16-31: K 16..31 (v0..7). Dword = (K_even, K_odd).
// Phase 2: per (t,d) accumulate over taps*c-chunks with bf16 hi/lo WMMA.
// ---------------------------------------------------------------------------
__global__ __launch_bounds__(256) void conv_wmma_kernel(
    const float* __restrict__ x,
    const unsigned int* __restrict__ wsHi,
    const unsigned int* __restrict__ wsLo,
    float* __restrict__ out) {
  __shared__ __attribute__((aligned(32))) unsigned int ldsA[2 * NP * 1024]; // hi | lo

  const int tid   = threadIdx.x;
  const int bbase = blockIdx.x * 16;

  // ---- Phase 1: build A fragments in LDS (each x element converted once) ----
  // flat slot s: v=s[2:0], lane=s[7:3], q=s[9:8], P=s[14:10]; lds dword idx == s.
  for (int it = 0; it < (NP * 1024) / 256; ++it) {
    const int s    = it * 256 + tid;
    const int v    = s & 7;
    const int lane = (s >> 3) & 31;
    const int q    = (s >> 8) & 3;
    const int P    = s >> 10;
    const int M    = lane & 15;
    const int half = lane >> 4;
    const int k0   = 2 * v + 16 * half;
    const int c    = q * 32 + k0;

    const float* xp = x + ((size_t)(bbase + M) * NC + c) * NLIN + P;
    const float a0 = xp[0];
    const float a1 = xp[NLIN];          // next c, same position
    unsigned int h0, l0, h1, l1;
    split2(a0, h0, l0);
    split2(a1, h1, l1);
    ldsA[s]             = (h1 << 16) | h0;
    ldsA[NP * 1024 + s] = (l1 << 16) | l0;
  }
  __syncthreads();

  // ---- Phase 2: WMMA accumulation ----
  const int w     = tid >> 5;       // wave id = o-tile
  const int lane  = tid & 31;
  const int obase = w * 16;
  const int N     = lane & 15;      // column within o-tile
  const int mtop  = (lane >> 4) * 8;
  float* outrow = out + ((size_t)(bbase + mtop) * NO + (obase + N)) * NSLOT;

#pragma unroll
  for (int d = 0; d < 6; ++d) {
    for (int t = 0; t < 9; ++t) {
      const int slot = t * 6 + d;
      if (t < TD[d]) {
        v8f acc = {};
#pragma unroll
        for (int j = TAP_START[d]; j < TAP_START[d + 1]; ++j) {
          const int P = 2 * t + TAP_P[j];
#pragma unroll
          for (int q = 0; q < 4; ++q) {
            const int abase = (P * 4 + q) * 256 + lane * 8;
            const v16bf ahi = *(const v16bf*)&ldsA[abase];
            const v16bf alo = *(const v16bf*)&ldsA[NP * 1024 + abase];
            const int wbase = (((j * 4 + q) * 8 + w) * 32 + lane) * 8;
            const v16bf bhi = *(const v16bf*)&wsHi[wbase];
            const v16bf blo = *(const v16bf*)&wsLo[wbase];
            // hi*hi + lo*hi + hi*lo  (drop lo*lo, ~2^-32): near-fp32 accuracy
            acc = __builtin_amdgcn_wmma_f32_16x16x32_bf16(
                false, ahi, false, bhi, (short)0, acc, false, false);
            acc = __builtin_amdgcn_wmma_f32_16x16x32_bf16(
                false, alo, false, bhi, (short)0, acc, false, false);
            acc = __builtin_amdgcn_wmma_f32_16x16x32_bf16(
                false, ahi, false, blo, (short)0, acc, false, false);
          }
        }
#pragma unroll
        for (int r = 0; r < 8; ++r)
          outrow[(size_t)r * (NO * NSLOT) + slot] = acc[r];
      } else {
        // t >= T_d: reference zero-pads
#pragma unroll
        for (int r = 0; r < 8; ++r)
          outrow[(size_t)r * (NO * NSLOT) + slot] = 0.0f;
      }
    }
  }
}

// ---------------------------------------------------------------------------
// Launch
// ---------------------------------------------------------------------------
extern "C" void kernel_launch(void* const* d_in, const int* in_sizes, int n_in,
                              void* d_out, int out_size, void* d_ws, size_t ws_size,
                              hipStream_t stream) {
  const float* x    = (const float*)d_in[0];
  const float* sums = (const float*)d_in[1];
  const float* f2   = (const float*)d_in[2];
  const float* f3   = (const float*)d_in[3];
  float* out = (float*)d_out;

  unsigned int* wsHi = (unsigned int*)d_ws;          // 229376 dwords
  unsigned int* wsLo = wsHi + WS_DW;                 // 229376 dwords (total ~1.8 MB)

  // 1) bake tap weights into WMMA B-fragment layout (bf16 hi/lo)
  prep_weights_kernel<<<WS_DW / 256, 256, 0, stream>>>(f2, f3, wsHi, wsLo);

  // 2) main WMMA conv: 1024 b-tiles of 16 rows
  conv_wmma_kernel<<<NB / 16, 256, 0, stream>>>(x, wsHi, wsLo, out);

  // 3) pass-through second output: sum_of_squares
  const size_t fwd_elems = (size_t)NB * NO * NSLOT;
  hipMemcpyAsync(out + fwd_elems, sums, (size_t)in_sizes[1] * sizeof(float),
                 hipMemcpyDeviceToDevice, stream);
}
